// CRF_16011638080097
// MI455X (gfx1250) — matrix operand order repaired and checked by reading, not compiled, not run on hardware
//
#include <hip/hip_runtime.h>
#include <hip/hip_bf16.h>

// CRF NLL on MI455X (gfx1250).
// Forward recurrence in exp-domain with max rescaling:
//   m = max_k alpha[k];  v = exp(alpha - m)   (f16, rescaled to [0,1])
//   alpha'[j] = logit[j] + m + log( (exp(transitions) @ v)[j] )
// -> v_wmma_f32_16x16x32_f16 per timestep, alpha register-resident in the
//    D-matrix layout, B fragments produced by one ds_store_b128 per lane.
// This revision: raw v_exp_f32/v_log_f32 (log2-domain, no libm guard code),
// 2 barriers/step (per-thread column-max reduce), vectorized logit staging.

typedef __attribute__((ext_vector_type(16))) _Float16 v16h;
typedef __attribute__((ext_vector_type(8)))  _Float16 v8h;
typedef __attribute__((ext_vector_type(8)))  float    v8f;
typedef __attribute__((ext_vector_type(4)))  float    v4f;

constexpr int Bn = 1024;
constexpr int Ln = 512;
constexpr int Tn = 50;     // tagset incl START/STOP
constexpr int TP = 64;     // padded tag dim
constexpr int LSP = 68;    // logit_s row stride (floats): bank-spread, 16B aligned
constexpr int START_T = 48;
constexpr int STOP_T  = 49;
constexpr float NEGV  = -10000.0f;
constexpr float LOG2E = 1.4426950408889634f;
constexpr float LN2   = 0.6931471805599453f;

__global__ void crf_zero(float* out) { out[0] = 0.0f; }

__global__ __launch_bounds__(128)
void crf_forward(const float* __restrict__ logits,
                 const float* __restrict__ trans,
                 const int*   __restrict__ lens,
                 float* __restrict__ out)
{
    __shared__ _Float16 Elds[TP * TP];       // exp(transitions) staging for A frags
    __shared__ _Float16 Vwmma[2 * 32 * 16];  // B fragments in register order [frag][lane][i]
    __shared__ float    logit_s[16 * LSP];   // [n][j], padded rows
    __shared__ float    pmaxT[16 * 8];       // [col][g] partial col maxes
    __shared__ int      slen[16];
    __shared__ float    alphaF[TP * 16];     // final alpha dump for partition reduce

    const int tid    = threadIdx.x;
    const int batch0 = blockIdx.x * 16;
    const int lane   = tid & 31;
    const int wt     = tid >> 5;        // wave row-tile 0..3
    const int lm     = lane & 15;       // col within tile (B/D) or row (A)
    const int lh     = (lane >> 4) & 1; // lane half: D rows r+8*lh

    // ---- build E = exp(transitions), padded 64x64, f16 ----
    for (int i = tid; i < TP * TP; i += 128) {
        int j = i >> 6, k = i & 63;
        float e = (j < Tn && k < Tn)
            ? __builtin_amdgcn_exp2f(trans[j * Tn + k] * LOG2E) : 0.0f;
        Elds[i] = (_Float16)e;
    }
    if (tid < 16) slen[tid] = lens[batch0 + tid];
    __syncthreads();

    // ---- persistent A fragments (16-bit A 16x32 layout, K chunks 0..31 / 32..63) ----
    v16h a0, a1;
    #pragma unroll
    for (int i = 0; i < 16; ++i) {
        int kk = 16 * lh + i;
        a0[i] = Elds[(16 * wt + lm) * TP + kk];
        a1[i] = Elds[(16 * wt + lm) * TP + 32 + kk];
    }

    // ---- alpha register-resident in D layout: row j = 16*wt + 8*lh + r, col lm ----
    float aR[8];
    #pragma unroll
    for (int r = 0; r < 8; ++r) {
        int j = 16 * wt + 8 * lh + r;
        aR[r] = (j == START_T) ? 0.0f : NEGV;
    }
    const int mylen = slen[lm];

    // producer slot for this lane's 8 exp values (contiguous i = 8*lh..8*lh+7)
    _Float16* vput = &Vwmma[(((wt >> 1) * 32) + (lm + 16 * (wt & 1))) * 16 + 8 * lh];

    // logit staging: this thread streams row bb, cols jb..jb+7 (index-clamped)
    const int bb = tid >> 3;
    const int jb = (tid & 7) * 8;
    const float* prow = logits + (size_t)(batch0 + bb) * Ln * Tn;
    int jc[8];
    #pragma unroll
    for (int e = 0; e < 8; ++e) jc[e] = (jb + e < Tn) ? (jb + e) : (Tn - 1);

    for (int t = 0; t < Ln; ++t) {
        // ---- phase A: per-lane partial column max; stream this step's logits ----
        float pm = aR[0];
        #pragma unroll
        for (int r = 1; r < 8; ++r) pm = fmaxf(pm, aR[r]);
        pmaxT[lm * 8 + wt * 2 + lh] = pm;

        float lg[8];
        #pragma unroll
        for (int e = 0; e < 8; ++e) lg[e] = prow[jc[e]];
        if (t + 1 < Ln)  // prefetch next step's chunk (global_prefetch_b8)
            __builtin_prefetch(prow + Tn + jc[0], 0, 0);
        prow += Tn;
        __syncthreads();

        // ---- phase C: every thread reduces its own column max; V = exp2 pack ----
        v4f px0 = *(const v4f*)&pmaxT[lm * 8];
        v4f px1 = *(const v4f*)&pmaxT[lm * 8 + 4];
        float mcl = fmaxf(fmaxf(fmaxf(px0[0], px0[1]), fmaxf(px0[2], px0[3])),
                          fmaxf(fmaxf(px1[0], px1[1]), fmaxf(px1[2], px1[3])));
        v8h vh;
        #pragma unroll
        for (int r = 0; r < 8; ++r)
            vh[r] = (_Float16)__builtin_amdgcn_exp2f((aR[r] - mcl) * LOG2E);
        *(v8h*)vput = vh;                       // one ds_store_b128

        v4f s0 = { lg[0], lg[1], lg[2], lg[3] };
        v4f s1 = { lg[4], lg[5], lg[6], lg[7] };
        *(v4f*)&logit_s[bb * LSP + jb]     = s0;   // ds_store_b128 x2
        *(v4f*)&logit_s[bb * LSP + jb + 4] = s1;
        __syncthreads();

        // ---- phase D: WMMA + raw log + masked (t < len) register update ----
        v16h b0 = *(const v16h*)&Vwmma[(0 * 32 + lane) * 16];  // ds_load_b128 x2
        v16h b1 = *(const v16h*)&Vwmma[(1 * 32 + lane) * 16];
        v8f acc = {};
        acc = __builtin_amdgcn_wmma_f32_16x16x32_f16(false, a0, false, b0,
                                                     (short)0, acc, false, false);
        acc = __builtin_amdgcn_wmma_f32_16x16x32_f16(false, a1, false, b1,
                                                     (short)0, acc, false, false);
        v4f lgt0 = *(const v4f*)&logit_s[lm * LSP + 16 * wt + 8 * lh];
        v4f lgt1 = *(const v4f*)&logit_s[lm * LSP + 16 * wt + 8 * lh + 4];
        const bool upd = (t < mylen);
        #pragma unroll
        for (int r = 0; r < 8; ++r) {
            int j = 16 * wt + 8 * lh + r;
            // raw v_log_f32 (log2): s==0 -> -inf, which propagates correctly
            float lv  = __builtin_amdgcn_logf(acc[r]) * LN2;
            float lgv = (r < 4) ? lgt0[r] : lgt1[r - 4];
            float nv  = lgv + mcl + lv;
            aR[r] = (upd && (j < Tn)) ? nv : aR[r];   // v_cndmask, no branches
        }
    }

    // ---- partition = logsumexp_j(alpha[j] + trans[STOP, j]) ----
    __syncthreads();
    #pragma unroll
    for (int r = 0; r < 8; ++r)
        alphaF[(16 * wt + 8 * lh + r) * 16 + lm] = aR[r];
    __syncthreads();
    if (tid < 16) {
        float mx = -3.4e38f;
        for (int j = 0; j < Tn; ++j)
            mx = fmaxf(mx, alphaF[j * 16 + tid] + trans[STOP_T * Tn + j]);
        float sum = 0.0f;
        for (int j = 0; j < Tn; ++j)
            sum += __builtin_amdgcn_exp2f(
                (alphaF[j * 16 + tid] + trans[STOP_T * Tn + j] - mx) * LOG2E);
        float part = mx + __builtin_amdgcn_logf(sum) * LN2;
        atomicAdd(out, part * (1.0f / Bn));
    }
}

// Emission + transition gold-path score (cheap, one thread per sequence)
__global__ __launch_bounds__(256)
void crf_score(const float* __restrict__ logits,
               const float* __restrict__ trans,
               const int*   __restrict__ labels,
               const int*   __restrict__ lens,
               float* __restrict__ out)
{
    int b = blockIdx.x * blockDim.x + threadIdx.x;
    if (b >= Bn) return;
    int len = lens[b];
    const int*   lab = labels + (size_t)b * Ln;
    const float* lg  = logits + (size_t)b * Ln * Tn;

    float em = 0.0f;
    for (int t = 0; t < Ln; ++t)
        if (t < len) em += lg[t * Tn + lab[t]];

    int prev = START_T;
    float ts = 0.0f;
    for (int tp = 0; tp <= Ln; ++tp) {
        int p = tp + 1;
        int nxt = (p <= Ln && p <= len) ? lab[p - 1] : STOP_T;
        if (tp <= len) ts += trans[nxt * Tn + prev];
        prev = nxt;
    }
    atomicAdd(out, -(em + ts) * (1.0f / Bn));
}

extern "C" void kernel_launch(void* const* d_in, const int* in_sizes, int n_in,
                              void* d_out, int out_size, void* d_ws, size_t ws_size,
                              hipStream_t stream) {
    const float* logits = (const float*)d_in[0];
    const float* trans  = (const float*)d_in[1];
    const int*   labels = (const int*)d_in[2];
    const int*   lens   = (const int*)d_in[3];
    float* out = (float*)d_out;

    crf_zero<<<1, 1, 0, stream>>>(out);
    crf_forward<<<Bn / 16, 128, 0, stream>>>(logits, trans, lens, out);
    crf_score<<<(Bn + 255) / 256, 256, 0, stream>>>(logits, trans, labels, lens, out);
}
